// OrthoT1d_34291018891867
// MI455X (gfx1250) — matrix operand order repaired and checked by reading, not compiled, not run on hardware
//
#include <hip/hip_runtime.h>

// ---------------------------------------------------------------------------
// OrthoT1d for MI455X (gfx1250, wave32, WMMA)
//
//   out = x @ A,  A = H_1 H_2 ... H_256,  H_i = I - 2 v_i v_i^T / ||v_i||^2
//
// Pipeline (6 launches):
//  1) build_blocks (16 WGs, parallel): compact-WY per 16-reflector block:
//       w_j = beta_j (v_j - W (Y^T v_j));  Q_b = I - W Y^T
//     Each Q_b written as fp32 row-major AND bf16 hi/lo packed in the WMMA
//     B-operand lane layout (so it can be a left or right operand later).
//  2) tree_gemm x4: A = ((Q0 Q1)(Q2 Q3))... binary-tree product, bf16x2 WMMA
//     GEMMs (4 WMMAs/K-chunk for extra accuracy), each output written in
//     both forms again. 8 -> 4 -> 2 -> 1 products.
//  3) ortho_gemm: HBM-roofline streaming GEMM (~268 MB @ 23.3 TB/s ~ 11.5us):
//     16 waves/block, B operand (hi+lo, K=256) register-resident per wave
//     (packed A is L2-resident); x tile split to bf16 hi/lo once per block
//     with trunc+v_perm packing into LDS (WMMA A-operand layout), double
//     buffered, 3 WMMAs per K-chunk (xh*Ah + xl*Ah + xh*Al). x loads and
//     out stores are non-temporal (single-use streams; don't pollute L2).
//
// Workspace: 31 nodes x (256KB fp32 + 256KB packed) ~= 15.5 MB.
// ---------------------------------------------------------------------------

typedef __attribute__((ext_vector_type(16))) __bf16 v16bf;
typedef __attribute__((ext_vector_type(8)))  float  v8f;
typedef __attribute__((ext_vector_type(4)))  float  v4f;

#define N_FEAT   256
#define KC_COUNT 8          // 256 / 32 K-chunks
#define NODE_SZ  65536      // 256*256 elements per tree node

static __device__ __forceinline__ unsigned short f2bf_rne(float f) {
  unsigned u = __float_as_uint(f);
  unsigned r = u + 0x7FFFu + ((u >> 16) & 1u);
  return (unsigned short)(r >> 16);
}
static __device__ __forceinline__ float bf2f(unsigned short h) {
  return __uint_as_float(((unsigned)h) << 16);
}
// pack top-16-bits of two floats into one dword {bf16(b),bf16(a)} (truncate)
static __device__ __forceinline__ unsigned pack_hi2(float a, float b) {
  return __builtin_amdgcn_perm(__float_as_uint(b), __float_as_uint(a), 0x07060302u);
}
static __device__ __forceinline__ float residual(float f) {
  return f - __uint_as_float(__float_as_uint(f) & 0xFFFF0000u);
}
// packed B-operand index for element (k, n) of a 256x256 matrix
static __device__ __forceinline__ int bpack_idx(int k, int n) {
  const int krem = k & 31;
  const int lane = ((krem & 8) ? 16 : 0) + (n & 15);
  return (((n >> 4) * KC_COUNT + (k >> 5)) * 32 + lane) * 16 + ((krem & 7) | ((krem & 16) >> 1));
}

// ---------------------------------------------------------------------------
// Kernel 1: per-block compact-WY build + Q_b materialization. 16 WGs x 256.
// Thread t owns dimension (column) t. yreg[j] = v_{16b+j}[t] = Y^T[t][j].
// ---------------------------------------------------------------------------
__global__ void __launch_bounds__(256)
build_blocks(const float* __restrict__ V, float* __restrict__ F,
             unsigned short* __restrict__ PH, unsigned short* __restrict__ PL) {
  __shared__ float WT[256 * 20];   // W^T, row stride 20 floats (16B-aligned,
                                   // bank-staggered); WT[t*20+l] = w_l[t]
  __shared__ float u[16];
  __shared__ float beta[16];

  const int t    = threadIdx.x;
  const int b    = blockIdx.x;
  const int lane = t & 31;

  float yreg[16];
#pragma unroll
  for (int j = 0; j < 16; ++j)
    yreg[j] = V[(size_t)(16 * b + j) * N_FEAT + t];

  // beta_j = 2 / ||v_j||^2  (16 cross-thread reductions)
  if (t < 16) beta[t] = 0.f;
  __syncthreads();
#pragma unroll
  for (int j = 0; j < 16; ++j) {
    float p = yreg[j] * yreg[j];
    p += __shfl_xor(p, 1, 32);  p += __shfl_xor(p, 2, 32);
    p += __shfl_xor(p, 4, 32);  p += __shfl_xor(p, 8, 32);
    p += __shfl_xor(p, 16, 32);
    if (lane == 0) atomicAdd(&beta[j], p);
  }
  __syncthreads();
  if (t < 16) beta[t] = 2.0f / beta[t];
  __syncthreads();

  // w_j = beta_j (v_j - W (Y^T v_j)) -- 16 short sequential steps
  float wreg[16];
#pragma unroll
  for (int j = 0; j < 16; ++j) {
    if (t < 16) u[t] = 0.f;
    __syncthreads();
#pragma unroll
    for (int l = 0; l < 16; ++l) {
      if (l < j) {
        float p = yreg[l] * yreg[j];
        p += __shfl_xor(p, 1, 32);  p += __shfl_xor(p, 2, 32);
        p += __shfl_xor(p, 4, 32);  p += __shfl_xor(p, 8, 32);
        p += __shfl_xor(p, 16, 32);
        if (lane == 0) atomicAdd(&u[l], p);
      }
    }
    __syncthreads();
    float s = yreg[j];
#pragma unroll
    for (int l = 0; l < 16; ++l)
      if (l < j) s -= wreg[l] * u[l];
    const float w = beta[j] * s;
    wreg[j] = w;
    WT[t * 20 + j] = w;
    __syncthreads();   // protect u before next iteration's zeroing
  }

  // Q_b[r][c] = delta(r,c) - sum_l W[l][r] * Y[l][c]; this thread: c = t,
  // Y[l][c] = yreg[l]. Write fp32 + packed bf16 hi/lo (RNE).
  float* Fb = F + (size_t)b * NODE_SZ;
  unsigned short* PHb = PH + (size_t)b * NODE_SZ;
  unsigned short* PLb = PL + (size_t)b * NODE_SZ;
  for (int r = 0; r < N_FEAT; ++r) {
    const float4 w0 = *(const float4*)&WT[r * 20];
    const float4 w1 = *(const float4*)&WT[r * 20 + 4];
    const float4 w2 = *(const float4*)&WT[r * 20 + 8];
    const float4 w3 = *(const float4*)&WT[r * 20 + 12];
    float q = (r == t) ? 1.0f : 0.0f;
    q -= w0.x * yreg[0]  + w0.y * yreg[1]  + w0.z * yreg[2]  + w0.w * yreg[3];
    q -= w1.x * yreg[4]  + w1.y * yreg[5]  + w1.z * yreg[6]  + w1.w * yreg[7];
    q -= w2.x * yreg[8]  + w2.y * yreg[9]  + w2.z * yreg[10] + w2.w * yreg[11];
    q -= w3.x * yreg[12] + w3.y * yreg[13] + w3.z * yreg[14] + w3.w * yreg[15];
    Fb[(size_t)r * N_FEAT + t] = q;
    const int idx = bpack_idx(r, t);
    const unsigned short hi = f2bf_rne(q);
    PHb[idx] = hi;
    PLb[idx] = f2bf_rne(q - bf2f(hi));
  }
}

// ---------------------------------------------------------------------------
// Kernel 2: one tree level. Product i = node(2i) @ node(2i+1).
// grid.x = nproducts*16 (16 M-tiles each), 512 threads = 16 waves (all N).
// ---------------------------------------------------------------------------
__global__ void __launch_bounds__(512)
tree_gemm(const float* __restrict__ srcF,
          const unsigned short* __restrict__ srcPH,
          const unsigned short* __restrict__ srcPL,
          float* __restrict__ dstF,
          unsigned short* __restrict__ dstPH,
          unsigned short* __restrict__ dstPL) {
  __shared__ __align__(32) unsigned short lh[KC_COUNT * 32 * 16];
  __shared__ __align__(32) unsigned short ll[KC_COUNT * 32 * 16];

  const int tid  = threadIdx.x;
  const int wave = tid >> 5;
  const int lane = tid & 31;
  const int prod = blockIdx.x >> 4;
  const int mt   = blockIdx.x & 15;

  // right child: packed B operand, register-resident
  const v16bf* Bh = (const v16bf*)(srcPH + (size_t)(2 * prod + 1) * NODE_SZ);
  const v16bf* Bl = (const v16bf*)(srcPL + (size_t)(2 * prod + 1) * NODE_SZ);
  v16bf bhi[KC_COUNT], blo[KC_COUNT];
#pragma unroll
  for (int kc = 0; kc < KC_COUNT; ++kc) {
    const int idx = (wave * KC_COUNT + kc) * 32 + lane;
    bhi[kc] = Bh[idx];
    blo[kc] = Bl[idx];
  }

  // left child: stage one 16x256 fp32 tile into LDS as A operands (hi/lo)
  const float* xm = srcF + (size_t)(2 * prod) * NODE_SZ;
  {
    const int srow  = tid & 15;
    const int k8    = tid >> 4;
    const int skc   = k8 >> 2;
    const int slane = srow + ((k8 & 1) ? 16 : 0);
    const int sslot = (k8 & 2) ? 8 : 0;
    const int sdst  = (skc * 32 + slane) * 16 + sslot;
    const float* p = xm + (size_t)(mt * 16 + srow) * N_FEAT + k8 * 8;
    const float4 f0 = *(const float4*)(p);
    const float4 f1 = *(const float4*)(p + 4);
    uint4 h, l;
    h.x = pack_hi2(f0.x, f0.y);  h.y = pack_hi2(f0.z, f0.w);
    h.z = pack_hi2(f1.x, f1.y);  h.w = pack_hi2(f1.z, f1.w);
    l.x = pack_hi2(residual(f0.x), residual(f0.y));
    l.y = pack_hi2(residual(f0.z), residual(f0.w));
    l.z = pack_hi2(residual(f1.x), residual(f1.y));
    l.w = pack_hi2(residual(f1.z), residual(f1.w));
    *(uint4*)&lh[sdst] = h;
    *(uint4*)&ll[sdst] = l;
  }
  __syncthreads();

  const v16bf* oh = (const v16bf*)lh;
  const v16bf* ol = (const v16bf*)ll;
  v8f acc = {};
#pragma unroll
  for (int kc = 0; kc < KC_COUNT; ++kc) {
    const v16bf xh = oh[kc * 32 + lane];
    const v16bf xl = ol[kc * 32 + lane];
    acc = __builtin_amdgcn_wmma_f32_16x16x32_bf16(false, xh, false, bhi[kc], (short)0, acc, false, false);
    acc = __builtin_amdgcn_wmma_f32_16x16x32_bf16(false, xl, false, bhi[kc], (short)0, acc, false, false);
    acc = __builtin_amdgcn_wmma_f32_16x16x32_bf16(false, xh, false, blo[kc], (short)0, acc, false, false);
    acc = __builtin_amdgcn_wmma_f32_16x16x32_bf16(false, xl, false, blo[kc], (short)0, acc, false, false);
  }

  // write product: fp32 row-major + packed bf16 hi/lo (RNE)
  const int rowHalf = (lane < 16) ? 0 : 8;
  const int colOut  = wave * 16 + (lane & 15);
  float* Fd = dstF + (size_t)prod * NODE_SZ;
  unsigned short* PHd = dstPH + (size_t)prod * NODE_SZ;
  unsigned short* PLd = dstPL + (size_t)prod * NODE_SZ;
#pragma unroll
  for (int v = 0; v < 8; ++v) {
    const int k = mt * 16 + rowHalf + v;
    const float q = acc[v];
    Fd[(size_t)k * N_FEAT + colOut] = q;
    const int idx = bpack_idx(k, colOut);
    const unsigned short hi = f2bf_rne(q);
    PHd[idx] = hi;
    PLd[idx] = f2bf_rne(q - bf2f(hi));
  }
}

// ---------------------------------------------------------------------------
// Kernel 3: HBM-roofline streaming GEMM out = x @ A  (A = packed final node).
// ---------------------------------------------------------------------------
__global__ void __launch_bounds__(512)
ortho_gemm(const float* __restrict__ x,
           const unsigned short* __restrict__ BhiU,
           const unsigned short* __restrict__ BloU,
           float* __restrict__ out,
           int mtiles) {
  __shared__ __align__(32) unsigned short lds_h[2][KC_COUNT * 32 * 16];
  __shared__ __align__(32) unsigned short lds_l[2][KC_COUNT * 32 * 16];

  const int tid  = threadIdx.x;
  const int wave = tid >> 5;
  const int lane = tid & 31;

  const v16bf* Bh = (const v16bf*)BhiU;
  const v16bf* Bl = (const v16bf*)BloU;
  v16bf bhi[KC_COUNT], blo[KC_COUNT];
#pragma unroll
  for (int kc = 0; kc < KC_COUNT; ++kc) {
    const int idx = (wave * KC_COUNT + kc) * 32 + lane;
    bhi[kc] = Bh[idx];
    blo[kc] = Bl[idx];
  }

  const int srow  = tid & 15;
  const int k8    = tid >> 4;
  const int skc   = k8 >> 2;
  const int slane = srow + ((k8 & 1) ? 16 : 0);
  const int sslot = (k8 & 2) ? 8 : 0;
  const int sdst  = (skc * 32 + slane) * 16 + sslot;

  const int rowHalf = (lane < 16) ? 0 : 8;
  const int colOut  = wave * 16 + (lane & 15);
  const int stride  = (int)gridDim.x;

  // Non-temporal loads: x is a single-use stream; keep it out of L2.
  // (ext-vector v4f pointer: __builtin_nontemporal_load needs a real vector
  //  type, not HIP's struct-based float4.)
  auto load8 = [&](int mt, v4f& f0, v4f& f1) {
    const v4f* p = (const v4f*)(x + (size_t)(mt * 16 + srow) * N_FEAT + k8 * 8);
    f0 = __builtin_nontemporal_load(p);
    f1 = __builtin_nontemporal_load(p + 1);
  };
  auto stage = [&](int b, const v4f& f0, const v4f& f1) {
    uint4 h, l;
    h.x = pack_hi2(f0.x, f0.y);  h.y = pack_hi2(f0.z, f0.w);
    h.z = pack_hi2(f1.x, f1.y);  h.w = pack_hi2(f1.z, f1.w);
    l.x = pack_hi2(residual(f0.x), residual(f0.y));
    l.y = pack_hi2(residual(f0.z), residual(f0.w));
    l.z = pack_hi2(residual(f1.x), residual(f1.y));
    l.w = pack_hi2(residual(f1.z), residual(f1.w));
    *(uint4*)&lds_h[b][sdst] = h;
    *(uint4*)&lds_l[b][sdst] = l;
  };

  if (blockIdx.x >= (unsigned)mtiles) return;

  {
    v4f f0, f1;
    load8(blockIdx.x, f0, f1);
    stage(0, f0, f1);
  }
  __syncthreads();

  int cur = 0;
  for (int mt = blockIdx.x; mt < mtiles; mt += stride) {
    const int mtn = mt + stride;
    v4f nf0, nf1;
    const bool have_next = (mtn < mtiles);
    if (have_next) load8(mtn, nf0, nf1);

    const v16bf* oh = (const v16bf*)lds_h[cur];
    const v16bf* ol = (const v16bf*)lds_l[cur];
    v8f acc = {};
#pragma unroll
    for (int kc = 0; kc < KC_COUNT; ++kc) {
      const v16bf xh = oh[kc * 32 + lane];
      const v16bf xl = ol[kc * 32 + lane];
      acc = __builtin_amdgcn_wmma_f32_16x16x32_bf16(false, xh, false, bhi[kc], (short)0, acc, false, false);
      acc = __builtin_amdgcn_wmma_f32_16x16x32_bf16(false, xl, false, bhi[kc], (short)0, acc, false, false);
      acc = __builtin_amdgcn_wmma_f32_16x16x32_bf16(false, xh, false, blo[kc], (short)0, acc, false, false);
    }

    // Non-temporal stores: out is write-only; don't dirty L2.
    float* orow = out + (size_t)(mt * 16 + rowHalf) * N_FEAT + colOut;
#pragma unroll
    for (int v = 0; v < 8; ++v)
      __builtin_nontemporal_store(acc[v], orow + (size_t)v * N_FEAT);

    if (have_next) stage(cur ^ 1, nf0, nf1);
    __syncthreads();
    cur ^= 1;
  }
}

// ---------------------------------------------------------------------------
// Launch: 6 kernels. Workspace layout (31 tree nodes):
//   F  : fp32 nodes,  31 * 65536 floats   ( 7.75 MB)
//   PH : packed hi,   31 * 65536 ushorts  ( 3.875 MB)
//   PL : packed lo,   31 * 65536 ushorts  ( 3.875 MB)
// Node order: leaves 0..15, L1 16..23, L2 24..27, L3 28..29, final 30.
// ---------------------------------------------------------------------------
extern "C" void kernel_launch(void* const* d_in, const int* in_sizes, int n_in,
                              void* d_out, int out_size, void* d_ws, size_t ws_size,
                              hipStream_t stream) {
  const float* x = (const float*)d_in[0];   // [32, 4096, 256] fp32
  const float* V = (const float*)d_in[1];   // [256, 256] fp32
  float* out = (float*)d_out;

  float* F = (float*)d_ws;
  unsigned short* PH = (unsigned short*)(F + (size_t)31 * NODE_SZ);
  unsigned short* PL = PH + (size_t)31 * NODE_SZ;

  const int M      = in_sizes[0] / N_FEAT;  // 131072 rows
  const int mtiles = M / 16;                // 8192 M tiles

  build_blocks<<<16, 256, 0, stream>>>(V, F, PH, PL);

  const size_t L0 = 0, L1 = (size_t)16 * NODE_SZ, L2 = (size_t)24 * NODE_SZ,
               L3 = (size_t)28 * NODE_SZ, L4 = (size_t)30 * NODE_SZ;
  tree_gemm<<<8 * 16, 512, 0, stream>>>(F + L0, PH + L0, PL + L0,
                                        F + L1, PH + L1, PL + L1);
  tree_gemm<<<4 * 16, 512, 0, stream>>>(F + L1, PH + L1, PL + L1,
                                        F + L2, PH + L2, PL + L2);
  tree_gemm<<<2 * 16, 512, 0, stream>>>(F + L2, PH + L2, PL + L2,
                                        F + L3, PH + L3, PL + L3);
  tree_gemm<<<1 * 16, 512, 0, stream>>>(F + L3, PH + L3, PL + L3,
                                        F + L4, PH + L4, PL + L4);

  int grid = 512;
  if (grid > mtiles) grid = mtiles;
  ortho_gemm<<<grid, 512, 0, stream>>>(x, PH + L4, PL + L4, out, mtiles);
}
